// QuantumHydraHybridLayer_19112604467489
// MI455X (gfx1250) — matrix operand order repaired and checked by reading, not compiled, not run on hardware
//
#include <hip/hip_runtime.h>
#include <hip/hip_bf16.h>

// CDNA5 (gfx1250) wave32 WMMA types
typedef __attribute__((ext_vector_type(16))) _Float16 v16h;
typedef __attribute__((ext_vector_type(8)))  float    v8f;

#define TWO_PI_F 6.28318f   // matches 2.0 * 3.14159 in the reference

__device__ __forceinline__ v8f wmma_f16(v16h a, v16h b, v8f c) {
  // (neg_a, A, neg_b, B, c_mod, C, reuse_a, reuse_b)
  return __builtin_amdgcn_wmma_f32_16x16x32_f16(false, a, false, b, (short)0, c,
                                                false, false);
}

// Pack 8 consecutive f32 (two float4s) into 8 f16 slots of a fragment.
__device__ __forceinline__ void pack8(v16h& d, int base, float4 lo, float4 hi) {
  d[base + 0] = (_Float16)lo.x;  d[base + 1] = (_Float16)lo.y;
  d[base + 2] = (_Float16)lo.z;  d[base + 3] = (_Float16)lo.w;
  d[base + 4] = (_Float16)hi.x;  d[base + 5] = (_Float16)hi.y;
  d[base + 6] = (_Float16)hi.z;  d[base + 7] = (_Float16)hi.w;
}

// ---------------------------------------------------------------------------
// Kernel 1: qp = sigmoid(x @ Wp^T + bp) * 2pi + qlcu_base     (256x96, K=128)
// One wave per 16x16 tile; K loop of 4 x (16x16x32) f16 WMMA, f32 accumulate.
// Per-lane fragment K indices are contiguous 8-float runs -> b128 loads.
// ---------------------------------------------------------------------------
__global__ __launch_bounds__(32) void qp_gemm_kernel(
    const float* __restrict__ x, const float* __restrict__ Wp,
    const float* __restrict__ bp, const float* __restrict__ qbase,
    float* __restrict__ qp) {
  const int lane   = threadIdx.x & 31;
  const int half   = lane >> 4;       // 0: lanes 0-15, 1: lanes 16-31
  const int lanelo = lane & 15;
  const int nb = blockIdx.x;          // 0..5  (96 cols)
  const int mb = blockIdx.y;          // 0..15 (256 rows)
  const int arow = mb * 16 + lanelo;  // A row held by this lane
  const int bcol = nb * 16 + lanelo;  // B column held by this lane
  const int aoff = half ? 8 : 0;      // A: K runs {aoff..aoff+7, aoff+16..aoff+23}
  const int boff = half ? 8 : 0;      // B: K runs {boff..boff+7, boff+16..boff+23}

  __builtin_prefetch(x + arow * 128, 0, 0);
  __builtin_prefetch(Wp + bcol * 128, 0, 0);

  v8f acc = {};
#pragma unroll
  for (int k0 = 0; k0 < 128; k0 += 32) {
    const float4* ap0 = reinterpret_cast<const float4*>(x + arow * 128 + k0 + aoff);
    const float4* ap1 = reinterpret_cast<const float4*>(x + arow * 128 + k0 + aoff + 16);
    const float4* bp0 = reinterpret_cast<const float4*>(Wp + bcol * 128 + k0 + boff);
    v16h a, bf;
    pack8(a, 0, ap0[0], ap0[1]);   // K = aoff..aoff+7
    pack8(a, 8, ap1[0], ap1[1]);   // K = aoff+16..aoff+23
    pack8(bf, 0, bp0[0], bp0[1]);  // K = boff..boff+7
    pack8(bf, 8, bp0[4], bp0[5]);  // K = boff+16..boff+23
    acc = wmma_f16(a, bf, acc);
  }
  // C/D layout: VGPR r -> row r + 8*half, lane -> col lanelo. Fused activation.
  const int n = nb * 16 + lanelo;
#pragma unroll
  for (int r = 0; r < 8; ++r) {
    const int m = mb * 16 + r + 8 * half;
    const float v  = acc[r] + bp[n];
    const float sg = __builtin_amdgcn_rcpf(1.0f + __expf(-v));  // fast sigmoid
    qp[m * 96 + n] = sg * TWO_PI_F + qbase[n];
  }
}

// ---------------------------------------------------------------------------
// Kernel 2: 12-qubit statevector simulation, one workgroup per batch sample.
// State (4096 complex amplitudes as interleaved float2 -> ds b64 traffic)
// lives entirely in LDS (32KB of 320KB/WGP); it never touches HBM. Wire
// permutations (shift/flip) are folded into measurement indexing, and
// flip(|0..0>)=|0..0> means m1/m2 share one simulation.
// ---------------------------------------------------------------------------
__device__ __forceinline__ void gate_ry(float2* st, int bb, float c, float s,
                                        int t) {
  const int bm = 1 << bb;
  for (int p = t; p < 2048; p += 256) {
    const int low = p & (bm - 1);
    const int i0  = ((p >> bb) << (bb + 1)) | low;
    const int i1  = i0 | bm;
    const float2 a0 = st[i0], a1 = st[i1];
    st[i0] = make_float2(c * a0.x - s * a1.x, c * a0.y - s * a1.y);
    st[i1] = make_float2(s * a0.x + c * a1.x, s * a0.y + c * a1.y);
  }
}

// exp(-i t/2 P) for P in {X(x)X, Y(x)Y}: a'[i] = c*a[i] - i*s*sgn(i)*a[i^mask]
__device__ __forceinline__ void gate_ising(float2* st, int b0, int mask,
                                           float c, float s, bool yy, int t) {
  const int bm = 1 << b0;
  const int om = mask ^ bm;  // the other coupled bit
  for (int p = t; p < 2048; p += 256) {
    const int low = p & (bm - 1);
    const int i   = ((p >> b0) << (b0 + 1)) | low;  // bit b0 of i is 0
    const int j   = i ^ mask;
    const float sg = yy ? ((i & om) ? 1.f : -1.f) : 1.f;  // YY: -1 when bits equal
    const float ss = s * sg;
    const float2 ai = st[i], aj = st[j];
    st[i] = make_float2(c * ai.x + ss * aj.y, c * ai.y - ss * aj.x);
    st[j] = make_float2(c * aj.x + ss * ai.y, c * aj.y - ss * ai.x);
  }
}

__global__ __launch_bounds__(256) void sim_kernel(
    const float* __restrict__ qp, float* __restrict__ m1,
    float* __restrict__ m2) {
  __shared__ float2 st[4096];
  __shared__ float acc[36];  // X[0..11], Y[12..23], Z[24..35] of psi
  const int t = threadIdx.x;
  const int b = blockIdx.x;

  for (int i = t; i < 4096; i += 256) st[i] = make_float2(i == 0 ? 1.f : 0.f, 0.f);
  if (t < 36) acc[t] = 0.f;
  __syncthreads();

  const float* p = qp + b * 96;
  int idx = 0;
  for (int layer = 0; layer < 2; ++layer) {
    for (int q = 0; q < 12; ++q) {  // RY sweep
      const float th = p[idx++];
      gate_ry(st, 11 - q, __cosf(0.5f * th), __sinf(0.5f * th), t);
      __syncthreads();
    }
    for (int q = 0; q < 12; ++q) {  // IsingXX ring
      const float th = p[idx++];
      const int b0 = 11 - q, b1 = 11 - ((q + 1) % 12);
      gate_ising(st, b0, (1 << b0) | (1 << b1), __cosf(0.5f * th),
                 __sinf(0.5f * th), false, t);
      __syncthreads();
    }
    for (int q = 0; q < 12; ++q) {  // RY sweep
      const float th = p[idx++];
      gate_ry(st, 11 - q, __cosf(0.5f * th), __sinf(0.5f * th), t);
      __syncthreads();
    }
    for (int q = 0; q < 12; ++q) {  // IsingYY ring
      const float th = p[idx++];
      const int b0 = 11 - q, b1 = 11 - ((q + 1) % 12);
      gate_ising(st, b0, (1 << b0) | (1 << b1), __cosf(0.5f * th),
                 __sinf(0.5f * th), true, t);
      __syncthreads();
    }
  }

  // Per-qubit <X>,<Y>,<Z> of psi via LDS float atomics (ds_add_f32).
  for (int q = 0; q < 12; ++q) {
    const int bb = 11 - q, bm = 1 << bb;
    float px = 0.f, py = 0.f, pz = 0.f;
    for (int p2 = t; p2 < 2048; p2 += 256) {
      const int low = p2 & (bm - 1);
      const int i0  = ((p2 >> bb) << (bb + 1)) | low;
      const int i1  = i0 | bm;
      const float2 a0 = st[i0], a1 = st[i1];
      px += a0.x * a1.x + a0.y * a1.y;   // Re(conj(a0)*a1)
      py += a0.x * a1.y - a0.y * a1.x;   // Im(conj(a0)*a1)
      pz += (a0.x * a0.x + a0.y * a0.y) - (a1.x * a1.x + a1.y * a1.y);
    }
    atomicAdd(&acc[q], 2.f * px);
    atomicAdd(&acc[12 + q], 2.f * py);
    atomicAdd(&acc[24 + q], pz);
  }
  __syncthreads();

  // m1: measured wire i of shift(psi) = orig qubit (i+1)%12
  // m2: measured wire i of flip(shift(psi)) = orig qubit (12-i)%12
  if (t < 12) {
    const int i = t;
    const int qa = (i + 1) % 12, qb = (12 - i) % 12;
    m1[b * 36 + i]      = acc[qa];
    m1[b * 36 + 12 + i] = acc[12 + qa];
    m1[b * 36 + 24 + i] = acc[24 + qa];
    m2[b * 36 + i]      = acc[qb];
    m2[b * 36 + 12 + i] = acc[12 + qb];
    m2[b * 36 + 24 + i] = acc[24 + qb];
  }
}

// ---------------------------------------------------------------------------
// Kernel 3: m3 in closed form. _qd on |0..0> is a product state:
// per qubit |v> = Rz(c)Ry(b)Rx(a)|0>; expectations from a single 2-vector.
// ---------------------------------------------------------------------------
__global__ __launch_bounds__(32) void m3_kernel(const float* __restrict__ qd,
                                                float* __restrict__ m3) {
  const int q = threadIdx.x;
  if (q >= 12) return;
  const float a = qd[q], bb = qd[12 + q], c = qd[24 + q];
  const float ca = __cosf(0.5f * a), sa = __sinf(0.5f * a);
  const float cb = __cosf(0.5f * bb), sb = __sinf(0.5f * bb);
  // Rx(a)|0> = (ca, -i*sa)
  float r0 = ca, i0 = 0.f, r1 = 0.f, i1 = -sa;
  // Ry(b): real rotation
  const float nr0 = cb * r0 - sb * r1, ni0 = cb * i0 - sb * i1;
  const float nr1 = sb * r0 + cb * r1, ni1 = sb * i0 + cb * i1;
  // Rz(c): v0 *= e^{-ic/2}, v1 *= e^{+ic/2}
  const float cc = __cosf(0.5f * c), sc = __sinf(0.5f * c);
  r0 = nr0 * cc + ni0 * sc;  i0 = ni0 * cc - nr0 * sc;
  r1 = nr1 * cc - ni1 * sc;  i1 = ni1 * cc + nr1 * sc;
  m3[q]      = 2.f * (r0 * r1 + i0 * i1);
  m3[12 + q] = 2.f * (r0 * i1 - i0 * r1);
  m3[24 + q] = (r0 * r0 + i0 * i0) - (r1 * r1 + i1 * i1);
}

// ---------------------------------------------------------------------------
// Kernel 4: fused head. yc = a1*(m1@W1^T+b1) + a2*(m2@W2^T+b2) + a3*(m3@W3^T+b3)
// WMMA GEMMs with |w|-normalized scales folded into the B fragments; the
// batch-constant m3 term + biases folded into the epilogue. K=36 -> padded 64.
// ---------------------------------------------------------------------------
__global__ __launch_bounds__(32) void head_gemm_kernel(
    const float* __restrict__ m1, const float* __restrict__ m2,
    const float* __restrict__ m3, const float* __restrict__ W1,
    const float* __restrict__ b1, const float* __restrict__ W2,
    const float* __restrict__ b2, const float* __restrict__ W3,
    const float* __restrict__ b3, const float* __restrict__ w1,
    const float* __restrict__ w2, const float* __restrict__ w3,
    float* __restrict__ yc) {
  const int lane   = threadIdx.x & 31;
  const int half   = lane >> 4;
  const int lanelo = lane & 15;
  const int nb = blockIdx.x;  // 0..2 (48 padded cols)
  const int mb = blockIdx.y;  // 0..15
  const float aw1 = fabsf(w1[0]), aw2 = fabsf(w2[0]), aw3 = fabsf(w3[0]);
  const float wsum = aw1 + aw2 + aw3;
  const float a1 = aw1 / wsum, a2 = aw2 / wsum, a3 = aw3 / wsum;

  const int arow = mb * 16 + lanelo;
  const int n    = nb * 16 + lanelo;

  v8f acc = {};
#pragma unroll
  for (int srcsel = 0; srcsel < 2; ++srcsel) {
    const float* A     = srcsel ? m2 : m1;
    const float* W     = srcsel ? W2 : W1;
    const float  scale = srcsel ? a2 : a1;
#pragma unroll
    for (int k0 = 0; k0 < 64; k0 += 32) {
      v16h af, bf;
#pragma unroll
      for (int v = 0; v < 8; ++v) {
        const int ka = k0 + (half ? 8 : 0) + ((v < 4) ? 2 * v : 16 + 2 * (v - 4));
        af[2 * v]     = (_Float16)((ka < 36)     ? A[arow * 36 + ka]     : 0.f);
        af[2 * v + 1] = (_Float16)((ka + 1 < 36) ? A[arow * 36 + ka + 1] : 0.f);
        const int kb = k0 + (half ? 8 : 0) + ((v < 4) ? 2 * v : 16 + 2 * (v - 4));
        const bool cn = (n < 36);
        bf[2 * v]     = (_Float16)((cn && kb < 36)     ? scale * W[n * 36 + kb]     : 0.f);
        bf[2 * v + 1] = (_Float16)((cn && kb + 1 < 36) ? scale * W[n * 36 + kb + 1] : 0.f);
      }
      acc = wmma_f16(af, bf, acc);
    }
  }
  // Batch-constant per-column term: a1*b1 + a2*b2 + a3*(m3.W3^T + b3)
  float cterm = 0.f;
  if (n < 36) {
    float d3 = 0.f;
#pragma unroll 4
    for (int k = 0; k < 36; ++k) d3 += m3[k] * W3[n * 36 + k];
    cterm = a1 * b1[n] + a2 * b2[n] + a3 * (d3 + b3[n]);
  }
#pragma unroll
  for (int r = 0; r < 8; ++r) {
    const int m = mb * 16 + r + 8 * half;
    if (n < 36) yc[m * 36 + n] = acc[r] + cterm;
  }
}

// ---------------------------------------------------------------------------
// Kernel 5: out = yc @ Wout^T + bout   (256x36 . 36 -> 256x1)
// ---------------------------------------------------------------------------
__global__ __launch_bounds__(256) void out_kernel(
    const float* __restrict__ yc, const float* __restrict__ Wout,
    const float* __restrict__ bout, float* __restrict__ out) {
  const int bidx = threadIdx.x;
  if (bidx >= 256) return;
  float s = bout[0];
#pragma unroll 4
  for (int j = 0; j < 36; ++j) s += yc[bidx * 36 + j] * Wout[j];
  out[bidx] = s;
}

// ---------------------------------------------------------------------------
extern "C" void kernel_launch(void* const* d_in, const int* in_sizes, int n_in,
                              void* d_out, int out_size, void* d_ws,
                              size_t ws_size, hipStream_t stream) {
  (void)in_sizes; (void)n_in; (void)out_size; (void)ws_size;
  const float* x     = (const float*)d_in[0];
  const float* Wp    = (const float*)d_in[1];
  const float* bp    = (const float*)d_in[2];
  const float* qbase = (const float*)d_in[3];
  const float* qd    = (const float*)d_in[4];
  const float* W1    = (const float*)d_in[5];
  const float* b1    = (const float*)d_in[6];
  const float* W2    = (const float*)d_in[7];
  const float* b2    = (const float*)d_in[8];
  const float* W3    = (const float*)d_in[9];
  const float* b3    = (const float*)d_in[10];
  const float* w1    = (const float*)d_in[11];
  const float* w2    = (const float*)d_in[12];
  const float* w3    = (const float*)d_in[13];
  const float* Wout  = (const float*)d_in[14];
  const float* bout  = (const float*)d_in[15];

  float* wsf = (float*)d_ws;
  float* qp = wsf;              // 256*96  = 24576 floats
  float* m1 = qp + 24576;       // 256*36  =  9216
  float* m2 = m1 + 9216;        // 256*36  =  9216
  float* m3 = m2 + 9216;        // 36 (pad to 64)
  float* yc = m3 + 64;          // 256*36  =  9216   -> ~208 KB total

  qp_gemm_kernel<<<dim3(6, 16), 32, 0, stream>>>(x, Wp, bp, qbase, qp);
  sim_kernel<<<256, 256, 0, stream>>>(qp, m1, m2);
  m3_kernel<<<1, 32, 0, stream>>>(qd, m3);
  head_gemm_kernel<<<dim3(3, 16), 32, 0, stream>>>(m1, m2, m3, W1, b1, W2, b2,
                                                   W3, b3, w1, w2, w3, yc);
  out_kernel<<<1, 256, 0, stream>>>(yc, Wout, bout, (float*)d_out);
}
